// P2AffinityAggregation_85152021611056
// MI455X (gfx1250) — compile-verified, matrix operand order checked by reading, not run on hardware
//
#include <hip/hip_runtime.h>
#include <hip/hip_bf16.h>
#include <float.h>

// ---------------- problem constants ----------------
#define N_NODES 3072
#define D_FEAT  128
#define E_CNT   98304
#define ET_CNT  101376   // E + N (self loops)
#define HID     512
#define LAT     32
#define LP_ITERS 30
#define V2C     3.5f

typedef __bf16 bf16_t;
typedef __attribute__((ext_vector_type(16))) __bf16 v16bf;
typedef __attribute__((ext_vector_type(8)))  float  v8f;
typedef __attribute__((ext_vector_type(2)))  float  v2f;

// ---------------- device helpers ----------------
__device__ __forceinline__ void edge_nodes(const int* __restrict__ ei, int e, int& rn, int& cn) {
    if (e < E_CNT) { rn = ei[e]; cn = ei[E_CNT + e]; }
    else           { rn = cn = e - E_CNT; }
}

// bf16 16x32 A/B fragment: lane holds row (lane&15); lanes 0-15: K=0..7,16..23 ; lanes 16-31: K=8..15,24..31
__device__ __forceinline__ v16bf load_frag_bf16(const bf16_t* __restrict__ p, int lane) {
    int kb = (lane & 16) ? 8 : 0;
    v16bf f;
#pragma unroll
    for (int i = 0; i < 8; ++i) f[i]     = p[kb + i];
#pragma unroll
    for (int i = 0; i < 8; ++i) f[8 + i] = p[kb + 16 + i];
    return f;
}

// f32 16x16x4 WMMA wrapper — called unconditionally; if the builtin is missing on this
// toolchain the compiler will say so explicitly ("use of unknown builtin").
__device__ __forceinline__ v8f wmma_f32x4(v2f a, v2f b, v8f c) {
    return __builtin_amdgcn_wmma_f32_16x16x4_f32(false, a, false, b, (short)0, c, false, false);
}

__device__ __forceinline__ float block_reduce_sum_256(float v) {
    __shared__ float sm[256];
    int t = threadIdx.x;
    sm[t] = v; __syncthreads();
    for (int s = 128; s > 0; s >>= 1) { if (t < s) sm[t] += sm[t + s]; __syncthreads(); }
    float r = sm[0]; __syncthreads();
    return r;
}

__device__ __forceinline__ void atomicMaxF(float* addr, float val) {
    unsigned* ua = (unsigned*)addr;
    unsigned old = *ua;
    while (true) {
        float f = __uint_as_float(old);
        if (f >= val) break;
        unsigned assumed = old;
        old = atomicCAS(ua, assumed, __float_as_uint(val));
        if (old == assumed) break;
    }
}

// ---------------- utility kernels ----------------
__global__ void k_fill_f32(float* p, float v, size_t n) {
    size_t i = (size_t)blockIdx.x * blockDim.x + threadIdx.x;
    size_t stride = (size_t)gridDim.x * blockDim.x;
    for (; i < n; i += stride) p[i] = v;
}
__global__ void k_fill_i32(int* p, int v, size_t n) {
    size_t i = (size_t)blockIdx.x * blockDim.x + threadIdx.x;
    size_t stride = (size_t)gridDim.x * blockDim.x;
    for (; i < n; i += stride) p[i] = v;
}
__global__ void k_cvt_bf16(const float* __restrict__ src, bf16_t* __restrict__ dst, size_t n) {
    size_t i = (size_t)blockIdx.x * blockDim.x + threadIdx.x;
    size_t stride = (size_t)gridDim.x * blockDim.x;
    for (; i < n; i += stride) dst[i] = (bf16_t)src[i];
}
// W [K,Ncols] row-major f32 -> WT [Ncols,K] bf16
__global__ void k_transpose_w(const float* __restrict__ W, bf16_t* __restrict__ WT, int K, int Ncols) {
    size_t idx = (size_t)blockIdx.x * blockDim.x + threadIdx.x;
    size_t n = (size_t)K * Ncols;
    size_t stride = (size_t)gridDim.x * blockDim.x;
    for (; idx < n; idx += stride) {
        int k = (int)(idx / Ncols), c = (int)(idx % Ncols);
        WT[(size_t)c * K + k] = (bf16_t)W[idx];
    }
}

// ---------------- generic bf16 WMMA GEMM: out = act(A[M,K] @ BT[N,K]^T + bias) ----------------
// one wave per 16x16 output tile; grid = (M/16, N/16); block = 32
template<bool RELU, bool OUT_BF16>
__global__ void k_gemm_bf16(const bf16_t* __restrict__ A, int lda,
                            const bf16_t* __restrict__ BT, int ldb,
                            const float* __restrict__ bias,
                            void* __restrict__ out, int ldo, int K) {
    int lane = threadIdx.x;
    int m0 = blockIdx.x * 16, n0 = blockIdx.y * 16;
    const bf16_t* arow = A  + (size_t)(m0 + (lane & 15)) * lda;
    const bf16_t* brow = BT + (size_t)(n0 + (lane & 15)) * ldb;
    v8f acc0 = {}, acc1 = {};
    for (int k0 = 0; k0 < K; k0 += 64) {
        v16bf a = load_frag_bf16(arow + k0, lane);
        v16bf b = load_frag_bf16(brow + k0, lane);
        acc0 = __builtin_amdgcn_wmma_f32_16x16x32_bf16(false, a, false, b, (short)0, acc0, false, false);
        if (k0 + 32 < K) {
            v16bf a2 = load_frag_bf16(arow + k0 + 32, lane);
            v16bf b2 = load_frag_bf16(brow + k0 + 32, lane);
            acc1 = __builtin_amdgcn_wmma_f32_16x16x32_bf16(false, a2, false, b2, (short)0, acc1, false, false);
        }
    }
    v8f acc = acc0 + acc1;
    int g = lane >> 4, n = lane & 15;
    float bv = bias ? bias[n0 + n] : 0.f;
#pragma unroll
    for (int r = 0; r < 8; ++r) {
        float v = acc[r] + bv;
        if (RELU) v = fmaxf(v, 0.f);
        size_t idx = (size_t)(m0 + 8 * g + r) * ldo + (n0 + n);
        if (OUT_BF16) ((bf16_t*)out)[idx] = (bf16_t)v;
        else          ((float*)out)[idx]  = v;
    }
}

// ---------------- encoder GEMM1, gather fused into A-load: pair = [x[row] | x[col]] ----------------
__global__ void k_gemm_encoder1(const bf16_t* __restrict__ xb,   // [N_NODES, 128]
                                const int* __restrict__ ei,
                                const bf16_t* __restrict__ BT,   // [512, 256]
                                const float* __restrict__ bias,  // [512]
                                bf16_t* __restrict__ out) {      // [ET, 512]
    int lane = threadIdx.x;
    int m0 = blockIdx.x * 16, n0 = blockIdx.y * 16;
    int eA = m0 + (lane & 15);
    int rn, cn; edge_nodes(ei, eA, rn, cn);
    const bf16_t* xr = xb + (size_t)rn * D_FEAT;
    const bf16_t* xc = xb + (size_t)cn * D_FEAT;
    const bf16_t* brow = BT + (size_t)(n0 + (lane & 15)) * 256;
    v8f acc0 = {}, acc1 = {};
    for (int k0 = 0; k0 < 256; k0 += 64) {
        const bf16_t* s1 = (k0 < 128) ? (xr + k0) : (xc + (k0 - 128));
        int k2 = k0 + 32;
        const bf16_t* s2 = (k2 < 128) ? (xr + k2) : (xc + (k2 - 128));
        v16bf a1 = load_frag_bf16(s1, lane);
        v16bf b1 = load_frag_bf16(brow + k0, lane);
        acc0 = __builtin_amdgcn_wmma_f32_16x16x32_bf16(false, a1, false, b1, (short)0, acc0, false, false);
        v16bf a2 = load_frag_bf16(s2, lane);
        v16bf b2 = load_frag_bf16(brow + k2, lane);
        acc1 = __builtin_amdgcn_wmma_f32_16x16x32_bf16(false, a2, false, b2, (short)0, acc1, false, false);
    }
    v8f acc = acc0 + acc1;
    int g = lane >> 4, n = lane & 15;
    float bv = bias[n0 + n];
#pragma unroll
    for (int r = 0; r < 8; ++r) {
        float v = fmaxf(acc[r] + bv, 0.f);
        out[(size_t)(m0 + 8 * g + r) * HID + (n0 + n)] = (bf16_t)v;
    }
}

// ---------------- decoder GEMM2 with fused recon-loss epilogue ----------------
// recon tile = A@B + bias ; sq-err against gathered f32 pair, reduced across the 16 N-lanes,
// atomicAdd into rl[e] (sum over 256 features).
__global__ void k_gemm_dec2_loss(const bf16_t* __restrict__ A,  // hd [ET,512]
                                 const bf16_t* __restrict__ BT, // [256,512]
                                 const float* __restrict__ bias,// [256]
                                 const float* __restrict__ x,   // f32 [N_NODES,128]
                                 const int* __restrict__ ei,
                                 float* __restrict__ rl) {      // [ET]
    int lane = threadIdx.x;
    int m0 = blockIdx.x * 16, n0 = blockIdx.y * 16;
    const bf16_t* arow = A  + (size_t)(m0 + (lane & 15)) * HID;
    const bf16_t* brow = BT + (size_t)(n0 + (lane & 15)) * HID;
    v8f acc0 = {}, acc1 = {};
    for (int k0 = 0; k0 < HID; k0 += 64) {
        v16bf a = load_frag_bf16(arow + k0, lane);
        v16bf b = load_frag_bf16(brow + k0, lane);
        acc0 = __builtin_amdgcn_wmma_f32_16x16x32_bf16(false, a, false, b, (short)0, acc0, false, false);
        v16bf a2 = load_frag_bf16(arow + k0 + 32, lane);
        v16bf b2 = load_frag_bf16(brow + k0 + 32, lane);
        acc1 = __builtin_amdgcn_wmma_f32_16x16x32_bf16(false, a2, false, b2, (short)0, acc1, false, false);
    }
    v8f acc = acc0 + acc1;
    int g = lane >> 4, n = lane & 15;
    int ng = n0 + n;
    float bv = bias[ng];
#pragma unroll
    for (int r = 0; r < 8; ++r) {
        int e = m0 + 8 * g + r;
        int rn, cn; edge_nodes(ei, e, rn, cn);
        float pv = (ng < D_FEAT) ? x[(size_t)rn * D_FEAT + ng]
                                 : x[(size_t)cn * D_FEAT + (ng - D_FEAT)];
        float d = acc[r] + bv - pv;
        float s = d * d;
        s += __shfl_xor(s, 1, 32);
        s += __shfl_xor(s, 2, 32);
        s += __shfl_xor(s, 4, 32);
        s += __shfl_xor(s, 8, 32);
        if (n == 0) atomicAdd(&rl[e], s);
    }
}

// ---------------- reparameterize + KL accumulation ----------------
__global__ void k_reparam(const float* __restrict__ muvar, const float* __restrict__ eps,
                          bf16_t* __restrict__ z, float* __restrict__ sums) {
    int idx = blockIdx.x * 256 + threadIdx.x;
    float kl = 0.f;
    if (idx < ET_CNT * LAT) {
        int e = idx >> 5, j = idx & 31;
        float mu = muvar[(size_t)e * 64 + j];
        float lv = muvar[(size_t)e * 64 + 32 + j];
        float zz = mu + eps[idx] * expf(0.5f * lv);
        z[idx] = (bf16_t)zz;
        kl = 1.f + lv - mu * mu - expf(lv);
    }
    float tot = block_reduce_sum_256(kl);
    if (threadIdx.x == 0) atomicAdd(&sums[1], tot);
}

// ---------------- affinities + recon-loss mean accumulation ----------------
__global__ void k_aff(const float* __restrict__ rl, float* __restrict__ aff, float* __restrict__ sums) {
    int e = blockIdx.x * 256 + threadIdx.x;
    float m = 0.f;
    if (e < ET_CNT) {
        m = rl[e] * (1.f / 256.f);
        aff[e] = expf(1.f / (1.f + V2C * m));
    }
    float tot = block_reduce_sum_256(m);
    if (threadIdx.x == 0) atomicAdd(&sums[0], tot);
}

// ---------------- scatter affinities into dense adjacency ----------------
__global__ void k_scatter_aff(const int* __restrict__ ei, const float* __restrict__ aff,
                              float* __restrict__ Ad) {
    int e = blockIdx.x * 256 + threadIdx.x;
    if (e >= ET_CNT) return;
    int rn, cn; edge_nodes(ei, e, rn, cn);
    atomicAdd(&Ad[(size_t)rn * N_NODES + cn], aff[e]);
}

__global__ void k_diag_one(float* __restrict__ LT) {
    int i = blockIdx.x * 256 + threadIdx.x;
    if (i < N_NODES) LT[(size_t)i * N_NODES + i] = 1.f;
}

// ---------------- LP dense GEMM (f32 WMMA): Agg = Ad @ L  (L given transposed as LT[n][k]) ----------
// Register-blocked: one wave computes a 32(M) x 64(N) block = 8 WMMA tiles.
// Per K-step (K=4): 2 A-fragment loads + 4 B-fragment loads feed 8 independent WMMAs
// (10.7 FLOP per L2 byte; dependence distance between same-acc WMMAs = 8 instructions).
// f32 16x16x4 fragment layout: lane holds row (lane&15); lanes 0-15: K=0,1 ; lanes 16-31: K=2,3.
__global__ void k_lp_gemm(const float* __restrict__ Ad, const float* __restrict__ LT,
                          float* __restrict__ Agg) {
    int lane = threadIdx.x;
    int m0 = blockIdx.x * 32;
    int n0 = blockIdx.y * 64;
    int kb = (lane & 16) ? 2 : 0;
    const float* a0r = Ad + (size_t)(m0 +      (lane & 15)) * N_NODES + kb;
    const float* a1r = Ad + (size_t)(m0 + 16 + (lane & 15)) * N_NODES + kb;
    const float* b0r = LT + (size_t)(n0 +      (lane & 15)) * N_NODES + kb;
    const float* b1r = LT + (size_t)(n0 + 16 + (lane & 15)) * N_NODES + kb;
    const float* b2r = LT + (size_t)(n0 + 32 + (lane & 15)) * N_NODES + kb;
    const float* b3r = LT + (size_t)(n0 + 48 + (lane & 15)) * N_NODES + kb;
    v8f acc0 = {}, acc1 = {}, acc2 = {}, acc3 = {};
    v8f acc4 = {}, acc5 = {}, acc6 = {}, acc7 = {};
#pragma unroll 2
    for (int k0 = 0; k0 < N_NODES; k0 += 4) {
        v2f a0 = *(const v2f*)(a0r + k0);
        v2f a1 = *(const v2f*)(a1r + k0);
        v2f b0 = *(const v2f*)(b0r + k0);
        v2f b1 = *(const v2f*)(b1r + k0);
        v2f b2 = *(const v2f*)(b2r + k0);
        v2f b3 = *(const v2f*)(b3r + k0);
        acc0 = wmma_f32x4(a0, b0, acc0);
        acc1 = wmma_f32x4(a0, b1, acc1);
        acc2 = wmma_f32x4(a0, b2, acc2);
        acc3 = wmma_f32x4(a0, b3, acc3);
        acc4 = wmma_f32x4(a1, b0, acc4);
        acc5 = wmma_f32x4(a1, b1, acc5);
        acc6 = wmma_f32x4(a1, b2, acc6);
        acc7 = wmma_f32x4(a1, b3, acc7);
    }
    // store: tile (i,j): rows m0+16i+8g+r, cols n0+16j+n
    int g = lane >> 4, n = lane & 15;
    v8f accs[8] = { acc0, acc1, acc2, acc3, acc4, acc5, acc6, acc7 };
#pragma unroll
    for (int i = 0; i < 2; ++i) {
#pragma unroll
        for (int j = 0; j < 4; ++j) {
            v8f c = accs[i * 4 + j];
            float* outp = Agg + (size_t)(m0 + 16 * i + 8 * g) * N_NODES + (n0 + 16 * j + n);
#pragma unroll
            for (int r = 0; r < 8; ++r) outp[(size_t)r * N_NODES] = c[r];
        }
    }
}

// ---------------- row-normalize Agg and write transposed into LT ----------------
__global__ void k_norm_T(const float* __restrict__ Agg, float* __restrict__ LT) {
    int r = blockIdx.x;
    const float* row = Agg + (size_t)r * N_NODES;
    float s = 0.f;
    for (int n = threadIdx.x; n < N_NODES; n += 256) s += row[n];
    float tot = block_reduce_sum_256(s);
    float inv = 1.f / (tot + 1e-8f);
    for (int n = threadIdx.x; n < N_NODES; n += 256)
        LT[(size_t)n * N_NODES + r] = row[n] * inv;
}

// ---------------- argmax over final Agg rows (first-max tiebreak) ----------------
__global__ void k_argmax(const float* __restrict__ Agg, int* __restrict__ cl,
                         float* __restrict__ out_cluster) {
    int r = blockIdx.x;
    const float* row = Agg + (size_t)r * N_NODES;
    float bv = -FLT_MAX; int bi = 0x7fffffff;
    for (int n = threadIdx.x; n < N_NODES; n += 256) {
        float v = row[n];
        if (v > bv || (v == bv && n < bi)) { bv = v; bi = n; }
    }
    __shared__ float sv[256];
    __shared__ int   si[256];
    int t = threadIdx.x;
    sv[t] = bv; si[t] = bi; __syncthreads();
    for (int s2 = 128; s2 > 0; s2 >>= 1) {
        if (t < s2) {
            if (sv[t + s2] > sv[t] || (sv[t + s2] == sv[t] && si[t + s2] < si[t])) {
                sv[t] = sv[t + s2]; si[t] = si[t + s2];
            }
        }
        __syncthreads();
    }
    if (t == 0) { cl[r] = si[0]; out_cluster[r] = (float)si[0]; }
}

// ---------------- segment-max pooling scatter ----------------
__global__ void k_pool_scatter(const float* __restrict__ x, const int* __restrict__ batch,
                               const int* __restrict__ cl, float* __restrict__ pooled,
                               int* __restrict__ used, int* __restrict__ cbmax) {
    int idx = blockIdx.x * 256 + threadIdx.x;
    if (idx >= N_NODES * D_FEAT) return;
    int i = idx >> 7, d = idx & 127;
    int c = cl[i];
    atomicMaxF(&pooled[(size_t)c * D_FEAT + d], x[idx]);
    if (d == 0) { used[c] = 1; atomicMax(&cbmax[c], batch[i]); }
}

__global__ void k_out_x(const float* __restrict__ pooled, const int* __restrict__ used,
                        float* __restrict__ out_x) {
    int idx = blockIdx.x * 256 + threadIdx.x;
    if (idx >= N_NODES * D_FEAT) return;
    out_x[idx] = used[idx >> 7] ? pooled[idx] : 0.f;
}

__global__ void k_out_misc(const int* __restrict__ used, const int* __restrict__ cbmax,
                           const float* __restrict__ sums, float* __restrict__ out_batch,
                           float* __restrict__ out_means) {
    int r = blockIdx.x * 256 + threadIdx.x;
    if (r < N_NODES) out_batch[r] = used[r] ? (float)cbmax[r] : 0.f;
    if (r == 0) {
        out_means[0] = sums[0] / (float)ET_CNT;            // mean recon loss
        out_means[1] = -0.5f * sums[1] / (float)ET_CNT;    // mean KL loss
    }
}

__global__ void k_adj_scatter(const int* __restrict__ ei, const int* __restrict__ cl,
                              float* __restrict__ adj) {
    int e = blockIdx.x * 256 + threadIdx.x;
    if (e >= ET_CNT) return;
    int rn, cn; edge_nodes(ei, e, rn, cn);
    adj[(size_t)cl[rn] * N_NODES + cl[cn]] = 1.0f;
}

// ---------------- host launch ----------------
static size_t alignup(size_t x) { return (x + 255) & ~(size_t)255; }

extern "C" void kernel_launch(void* const* d_in, const int* in_sizes, int n_in,
                              void* d_out, int out_size, void* d_ws, size_t ws_size,
                              hipStream_t stream) {
    const float* x    = (const float*)d_in[0];
    const int*   ei   = (const int*)d_in[1];
    const int*   batch= (const int*)d_in[2];
    const float* eps  = (const float*)d_in[3];
    const float* w_e1 = (const float*)d_in[4];
    const float* b_e1 = (const float*)d_in[5];
    const float* w_e2 = (const float*)d_in[6];
    const float* b_e2 = (const float*)d_in[7];
    const float* w_d1 = (const float*)d_in[8];
    const float* b_d1 = (const float*)d_in[9];
    const float* w_d2 = (const float*)d_in[10];
    const float* b_d2 = (const float*)d_in[11];

    // workspace layout
    char* ws = (char*)d_ws;
    size_t off = 0;
    auto take = [&](size_t bytes) { char* p = ws + off; off += alignup(bytes); return p; };
    bf16_t* XB    = (bf16_t*)take((size_t)N_NODES * D_FEAT * 2);
    bf16_t* WE1T  = (bf16_t*)take((size_t)HID * 2 * D_FEAT * 2);        // [512,256]
    bf16_t* WE2T  = (bf16_t*)take((size_t)2 * LAT * HID * 2);           // [64,512]
    bf16_t* WD1T  = (bf16_t*)take((size_t)HID * LAT * 2);               // [512,32]
    bf16_t* WD2T  = (bf16_t*)take((size_t)2 * D_FEAT * HID * 2);        // [256,512]
    bf16_t* H     = (bf16_t*)take((size_t)ET_CNT * HID * 2);            // reused as HD
    float*  MUVAR = (float*) take((size_t)ET_CNT * 64 * 4);
    bf16_t* Z     = (bf16_t*)take((size_t)ET_CNT * LAT * 2);
    float*  RL    = (float*) take((size_t)ET_CNT * 4);
    float*  AFF   = (float*) take((size_t)ET_CNT * 4);
    float*  AD    = (float*) take((size_t)N_NODES * N_NODES * 4);
    float*  LT    = (float*) take((size_t)N_NODES * N_NODES * 4);
    float*  AGG   = (float*) take((size_t)N_NODES * N_NODES * 4);
    float*  POOL  = (float*) take((size_t)N_NODES * D_FEAT * 4);
    int*    USED  = (int*)   take((size_t)N_NODES * 4);
    int*    CBMAX = (int*)   take((size_t)N_NODES * 4);
    int*    CL    = (int*)   take((size_t)N_NODES * 4);
    float*  SUMS  = (float*) take(256);

    // output layout (floats)
    float* out       = (float*)d_out;
    float* out_x     = out;
    float* out_adj   = out + (size_t)N_NODES * D_FEAT;
    float* out_batch = out_adj + (size_t)N_NODES * N_NODES;
    float* out_clust = out_batch + N_NODES;
    float* out_means = out_clust + N_NODES;

    // ---- init fills ----
    k_fill_f32<<<64, 256, 0, stream>>>(SUMS, 0.f, 64);
    k_fill_f32<<<512, 256, 0, stream>>>(RL, 0.f, (size_t)ET_CNT);
    k_fill_f32<<<4096, 256, 0, stream>>>(AD, 0.f, (size_t)N_NODES * N_NODES);
    k_fill_f32<<<4096, 256, 0, stream>>>(LT, 0.f, (size_t)N_NODES * N_NODES);
    k_fill_f32<<<4096, 256, 0, stream>>>(out_adj, 0.f, (size_t)N_NODES * N_NODES);
    k_fill_f32<<<2048, 256, 0, stream>>>(POOL, -FLT_MAX, (size_t)N_NODES * D_FEAT);
    k_fill_i32<<<16, 256, 0, stream>>>(USED, 0, (size_t)N_NODES);
    k_fill_i32<<<16, 256, 0, stream>>>(CBMAX, INT_MIN, (size_t)N_NODES);
    k_diag_one<<<(N_NODES + 255) / 256, 256, 0, stream>>>(LT);

    // ---- precision conversions / weight transposes ----
    k_cvt_bf16<<<1536, 256, 0, stream>>>(x, XB, (size_t)N_NODES * D_FEAT);
    k_transpose_w<<<512, 256, 0, stream>>>(w_e1, WE1T, 2 * D_FEAT, HID);   // [256,512]->[512,256]
    k_transpose_w<<<128, 256, 0, stream>>>(w_e2, WE2T, HID, 2 * LAT);      // [512,64]->[64,512]
    k_transpose_w<<<64, 256, 0, stream>>>(w_d1, WD1T, LAT, HID);           // [32,512]->[512,32]
    k_transpose_w<<<512, 256, 0, stream>>>(w_d2, WD2T, HID, 2 * D_FEAT);   // [512,256]->[256,512]

    const int MT = ET_CNT / 16;  // 6336 row tiles

    // ---- VAE forward ----
    k_gemm_encoder1<<<dim3(MT, HID / 16), 32, 0, stream>>>(XB, ei, WE1T, b_e1, H);
    k_gemm_bf16<false, false><<<dim3(MT, 64 / 16), 32, 0, stream>>>(H, HID, WE2T, HID, b_e2, MUVAR, 64, HID);
    k_reparam<<<(ET_CNT * LAT + 255) / 256, 256, 0, stream>>>(MUVAR, eps, Z, SUMS);
    k_gemm_bf16<true, true><<<dim3(MT, HID / 16), 32, 0, stream>>>(Z, LAT, WD1T, LAT, b_d1, H, HID, LAT);
    k_gemm_dec2_loss<<<dim3(MT, (2 * D_FEAT) / 16), 32, 0, stream>>>(H, WD2T, b_d2, x, ei, RL);
    k_aff<<<(ET_CNT + 255) / 256, 256, 0, stream>>>(RL, AFF, SUMS);

    // ---- label propagation ----
    k_scatter_aff<<<(ET_CNT + 255) / 256, 256, 0, stream>>>(ei, AFF, AD);
    for (int it = 0; it < LP_ITERS; ++it) {
        k_lp_gemm<<<dim3(N_NODES / 32, N_NODES / 64), 32, 0, stream>>>(AD, LT, AGG);
        if (it + 1 < LP_ITERS)
            k_norm_T<<<N_NODES, 256, 0, stream>>>(AGG, LT);
    }
    k_argmax<<<N_NODES, 256, 0, stream>>>(AGG, CL, out_clust);

    // ---- pooling + coarse outputs ----
    k_pool_scatter<<<(N_NODES * D_FEAT + 255) / 256, 256, 0, stream>>>(x, batch, CL, POOL, USED, CBMAX);
    k_out_x<<<(N_NODES * D_FEAT + 255) / 256, 256, 0, stream>>>(POOL, USED, out_x);
    k_out_misc<<<(N_NODES + 255) / 256, 256, 0, stream>>>(USED, CBMAX, SUMS, out_batch, out_means);
    k_adj_scatter<<<(ET_CNT + 255) / 256, 256, 0, stream>>>(ei, CL, out_adj);
}
